// DoubleTFlearningCNet_Transformer_704374637342
// MI455X (gfx1250) — compile-verified, matrix operand order checked by reading, not run on hardware
//
#include <hip/hip_runtime.h>
#include <hip/hip_bf16.h>

// ======================================================================
// DoubleTFlearningCNet forward for MI455X (gfx1250).
// Heavy convs run as bf16 WMMA implicit GEMM (v_wmma_f32_16x16x32_bf16,
// wave32, 16oc x 16pos tile per wave), templated on kernel size so all
// address math is shifts/constants. All gathers use uniform-base +
// 32-bit element indices (SADDR+voffset form); per-lane LDS tables hold
// precomputed reflected/padded pixel indices (one ds_load_b64 per pair);
// bf16 packing is a single v_perm_b32.
// Workspace requirement: ~1.15 GB in d_ws (layout in kernel_launch).
// ======================================================================

#define SP 65536      // 256*256 spatial positions
#define HW 256

typedef __bf16 bf16_t;
typedef bf16_t v16bf __attribute__((ext_vector_type(16)));
typedef unsigned short v16us __attribute__((ext_vector_type(16)));
typedef float v8f __attribute__((ext_vector_type(8)));

union BFrag { unsigned u32[8]; v16us u16; v16bf v; };

// truncating fp32->bf16 pack of two values into one dword: one v_perm_b32
__device__ __forceinline__ unsigned pack2(float lo, float hi) {
  // result bytes: [lo.b2, lo.b3, hi.b2, hi.b3]  (sel 0..3 = src1, 4..7 = src0)
  return __builtin_amdgcn_perm(__float_as_uint(hi), __float_as_uint(lo), 0x07060302u);
}

__device__ __forceinline__ int refl255(int i) {   // jnp 'reflect' pad (p<=5)
  i = i < 0 ? -i : i;
  return i > 255 ? 510 - i : i;
}

// ----------------------------------------------------------------------
// Implicit-GEMM conv via WMMA, templated on kernel size.
//   KS=11: reflect pad 5, K padded 121->128 per channel (ci = k>>7)
//   KS=3 : zero pad 1,   K padded   9->16  per channel (2 channels/chunk)
//   KS=1 : plain GEMM over channels
//   ROT  : weight addressed as rot180+transpose: wT[oc,ci,f] = w[ci*wD1+oc][120-f]
// out[b,oc,y,x] = clip(scale*(bias[oc] + sum), lo, hi)
// ----------------------------------------------------------------------
template<int KS, bool ROT>
__global__ __launch_bounds__(32)
void wmma_conv_t(const float* __restrict__ in, int inCtot, int inCoff,
                 const float* __restrict__ wgt, int wD1,
                 const float* __restrict__ bias,
                 float* __restrict__ out, int outCtot,
                 int Cin, int Cout,
                 float scale, int doClip, float clipLo, float clipHi)
{
  const int lane = threadIdx.x;
  const int l15  = lane & 15;
  const int kh8  = (lane >> 4) << 3;            // 0 or 8

  const int px = (blockIdx.x & 15) << 4;        // x tile base
  const int py = (blockIdx.x >> 4) & 255;       // row
  const int b  = blockIdx.x >> 12;              // batch
  const int ox = px + l15;                      // B column (N)
  const int oc = blockIdx.y * 16 + l15;         // A row (M)
  const int oc_c = oc < Cout ? oc : Cout - 1;   // clamped for addressing only

  v8f acc = {};

  if constexpr (KS == 1) {
    // ---- plain GEMM over channels (indices 32-bit, uniform bases) ------
    const unsigned chan0 = (unsigned)(b * inCtot + inCoff) << 16;
    const unsigned sp0   = (unsigned)(py * HW + ox);
    const unsigned wrow0 = (unsigned)oc_c * (unsigned)Cin;
    for (int k0 = 0; k0 < Cin; k0 += 32) {
      BFrag A, B;
#pragma unroll
      for (int j = 0; j < 8; ++j) {
        const int kb = ((j < 4) ? 0 : 16) + kh8 + ((j & 3) << 1);
        const int k0j = k0 + kb;
        const unsigned ka0 = (unsigned)((k0j     < Cin) ? k0j     : 0);
        const unsigned ka1 = (unsigned)((k0j + 1 < Cin) ? k0j + 1 : 0);
        float w0 = wgt[wrow0 + ka0]; if (k0j     >= Cin) w0 = 0.f;
        float w1 = wgt[wrow0 + ka1]; if (k0j + 1 >= Cin) w1 = 0.f;
        const float x0 = in[chan0 + (ka0 << 16) + sp0];
        const float x1 = in[chan0 + (ka1 << 16) + sp0];
        A.u32[j] = pack2(w0, w1);
        B.u32[j] = pack2(x0, x1);
      }
      acc = __builtin_amdgcn_wmma_f32_16x16x32_bf16(false, A.v, false, B.v,
                                                    (short)0, acc, false, false);
    }
  } else if constexpr (KS == 11) {
    // ---- 11x11 reflect-pad path ---------------------------------------
    __shared__ unsigned offT[32 * 128];
    unsigned* myT = offT + lane * 128;          // lane-private, no barrier needed
    for (int r = 0; r < 128; ++r) {
      const int rr = (r < 121) ? r : 0;
      const int ky = rr / 11;
      const int kx = rr - ky * 11;
      const int iy = refl255(py + ky - 5);
      const int ix = refl255(ox + kx - 5);
      myT[r] = (unsigned)(iy * HW + ix);        // element index within channel
    }
    unsigned chBase = (unsigned)(b * inCtot + inCoff) << 16;   // element idx
    unsigned wRow   = ROT ? (unsigned)oc_c * 121u
                          : (unsigned)oc_c * (unsigned)Cin * 121u;
    const unsigned wStride = ROT ? (unsigned)wD1 * 121u : 121u;
    for (int ci = 0; ci < Cin; ++ci) {
      __builtin_prefetch(&in[chBase + 65536u + myT[0]], 0, 1);  // next channel
#pragma unroll
      for (int rb = 0; rb < 128; rb += 32) {
        BFrag A, B;
#pragma unroll
        for (int j = 0; j < 8; ++j) {
          const int kb = ((j < 4) ? 0 : 16) + kh8 + ((j & 3) << 1);
          const int r0 = rb + kb, r1 = r0 + 1;
          const unsigned ra0 = (unsigned)((r0 < 121) ? (ROT ? 120 - r0 : r0) : 0);
          const unsigned ra1 = (unsigned)((r1 < 121) ? (ROT ? 120 - r1 : r1) : 0);
          float w0 = wgt[wRow + ra0]; if (r0 >= 121) w0 = 0.f;
          float w1 = wgt[wRow + ra1]; if (r1 >= 121) w1 = 0.f;
          const uint2 o2 = *(const uint2*)(myT + r0);          // ds_load_b64
          const float x0 = in[chBase + o2.x];
          const float x1 = in[chBase + o2.y];
          A.u32[j] = pack2(w0, w1);
          B.u32[j] = pack2(x0, x1);
        }
        acc = __builtin_amdgcn_wmma_f32_16x16x32_bf16(false, A.v, false, B.v,
                                                      (short)0, acc, false, false);
      }
      chBase += 65536u;
      wRow   += wStride;
    }
  } else {
    // ---- 3x3 zero-pad path (two channels per 32-chunk) ----------------
    __shared__ unsigned offT[32 * 16];
    unsigned* myT = offT + lane * 16;
    unsigned oobm = 0;                           // bit r: zero this B element
#pragma unroll
    for (int r = 0; r < 16; ++r) {
      const int rr = (r < 9) ? r : 0;
      const int ky = rr / 3;
      const int kx = rr - ky * 3;
      int iy = py + ky - 1, ix = ox + kx - 1;
      const bool bad = (r >= 9) | (iy < 0) | (iy > 255) | (ix < 0) | (ix > 255);
      if (bad) oobm |= (1u << r);
      iy = iy < 0 ? 0 : (iy > 255 ? 255 : iy);
      ix = ix < 0 ? 0 : (ix > 255 ? 255 : ix);
      myT[r] = (unsigned)(iy * HW + ix);
    }
    unsigned chBase = (unsigned)(b * inCtot + inCoff) << 16;
    unsigned wRow   = (unsigned)oc_c * (unsigned)Cin * 9u;
    for (int ci0 = 0; ci0 < Cin; ci0 += 2) {
      BFrag A, B;
#pragma unroll
      for (int j = 0; j < 8; ++j) {
        const int rl = kh8 + ((j & 3) << 1);     // rem within channel, [0,16)
        const unsigned wR = wRow   + ((j < 4) ? 0u : 9u);
        const unsigned cB = chBase + ((j < 4) ? 0u : 65536u);
        const unsigned ra0 = (unsigned)(rl     < 9 ? rl     : 8);
        const unsigned ra1 = (unsigned)(rl + 1 < 9 ? rl + 1 : 8);
        const float w0 = wgt[wR + ra0];
        const float w1 = wgt[wR + ra1];
        const uint2 o2 = *(const uint2*)(myT + rl);
        float x0 = in[cB + o2.x]; if ((oobm >> rl)       & 1u) x0 = 0.f;
        float x1 = in[cB + o2.y]; if ((oobm >> (rl + 1)) & 1u) x1 = 0.f;
        A.u32[j] = pack2(w0, w1);
        B.u32[j] = pack2(x0, x1);
      }
      acc = __builtin_amdgcn_wmma_f32_16x16x32_bf16(false, A.v, false, B.v,
                                                    (short)0, acc, false, false);
      chBase += 2u * 65536u;
      wRow   += 18u;
    }
  }

  // D layout: lanes 0-15 -> N=lane, M=r; lanes 16-31 -> N=lane-16, M=8+r
#pragma unroll
  for (int r = 0; r < 8; ++r) {
    const int occ = blockIdx.y * 16 + kh8 + r;
    if (occ < Cout) {
      float v = acc[r];
      if (bias) v += bias[occ];
      v *= scale;
      if (doClip) v = fminf(fmaxf(v, clipLo), clipHi);
      out[(((long long)b * outCtot + occ) << 16) + py * HW + px + l15] = v;
    }
  }
}

// ---------------- channel concat: xcat = [input(1ch), Wx(121ch)] ----------
__global__ void concat_kernel(const float* __restrict__ inp, const float* __restrict__ Wx,
                              float* __restrict__ xcat, long long n)
{
  long long i = (long long)blockIdx.x * blockDim.x + threadIdx.x;
  if (i >= n) return;
  int sp = (int)(i & 65535);
  long long bc = i >> 16;
  int c = (int)(bc % 122);
  int b = (int)(bc / 122);
  xcat[i] = (c == 0) ? inp[((long long)b << 16) + sp]
                     : Wx[(((long long)b * 121 + (c - 1)) << 16) + sp];
}

// ---------------- LayerNorm over channel dim per pixel; optional v-mul -----
__global__ void ln_c_kernel(const float* __restrict__ x, long long xB, long long xOff,
                            const float* __restrict__ vm, long long vB, long long vOff,
                            float* __restrict__ y, long long yB, long long yOff,
                            const float* __restrict__ w, const float* __restrict__ bb, int C)
{
  int pix = blockIdx.x * blockDim.x + threadIdx.x;
  if (pix >= 2 * SP) return;
  int b = pix >> 16, sp = pix & 65535;
  const float* xp = x + (long long)b * xB + xOff + sp;
  float s = 0.f, s2 = 0.f;
  for (int c = 0; c < C; ++c) { float v = xp[(long long)c << 16]; s += v; s2 += v * v; }
  float mu  = s / C;
  float var = s2 / C - mu * mu;
  float rstd = rsqrtf(var + 1e-5f);
  float* yp = y + (long long)b * yB + yOff + sp;
  if (vm) {
    const float* vp = vm + (long long)b * vB + vOff + sp;
    for (int c = 0; c < C; ++c) {
      float v = xp[(long long)c << 16];
      yp[(long long)c << 16] = vp[(long long)c << 16] * ((v - mu) * rstd * w[c] + bb[c]);
    }
  } else {
    for (int c = 0; c < C; ++c) {
      float v = xp[(long long)c << 16];
      yp[(long long)c << 16] = (v - mu) * rstd * w[c] + bb[c];
    }
  }
}

// ---------------- depthwise 3x3, zero pad, no bias -------------------------
__global__ void dw3x3_kernel(const float* __restrict__ x, const float* __restrict__ w,
                             float* __restrict__ y, int C, long long n)
{
  long long i = (long long)blockIdx.x * blockDim.x + threadIdx.x;
  if (i >= n) return;
  int sp = (int)(i & 65535);
  long long bc = i >> 16;
  int c = (int)(bc % C);
  int yy = sp >> 8, xx = sp & 255;
  const float* wp = w + c * 9;
  const float* xp = x + (bc << 16);
  float s = 0.f;
#pragma unroll
  for (int ky = 0; ky < 3; ++ky)
#pragma unroll
    for (int kx = 0; kx < 3; ++kx) {
      int iy = yy + ky - 1, ix = xx + kx - 1;
      if (iy >= 0 && iy < 256 && ix >= 0 && ix < 256)
        s += xp[iy * 256 + ix] * wp[ky * 3 + kx];
    }
  y[i] = s;
}

// ------- 8x8 patch circular conv:  o = irfft2(rfft2(q)*rfft2(k)) -----------
__global__ void patch_cconv_kernel(const float* __restrict__ h, float* __restrict__ o)
{
  int t = blockIdx.x;
  int patch = t & 1023;
  int rest  = t >> 10;
  int c = rest % 244;
  int b = rest / 244;
  int ph = patch >> 5, pw_ = patch & 31;
  int base = ph * 8 * 256 + pw_ * 8;
  int u = threadIdx.x >> 3, v = threadIdx.x & 7;
  __shared__ float Q[64], Kp[64];
  long long qb = (((long long)b * 732 + c) << 16) + base;
  long long kb = (((long long)b * 732 + c + 244) << 16) + base;
  Q[threadIdx.x]  = h[qb + u * 256 + v];
  Kp[threadIdx.x] = h[kb + u * 256 + v];
  __syncthreads();
  float s = 0.f;
#pragma unroll
  for (int ss = 0; ss < 8; ++ss)
#pragma unroll
    for (int tt = 0; tt < 8; ++tt)
      s += Q[ss * 8 + tt] * Kp[((u - ss) & 7) * 8 + ((v - tt) & 7)];
  o[(((long long)b * 732 + c) << 16) + base + u * 256 + v] = s;
}

// ---------------- misc elementwise ----------------------------------------
__global__ void add_inplace_kernel(float* __restrict__ a, const float* __restrict__ b, long long n)
{
  long long i = (long long)blockIdx.x * blockDim.x + threadIdx.x;
  if (i < n) a[i] += b[i];
}

__global__ void avgpool_kernel(const float* __restrict__ x, float* __restrict__ mean)
{
  __shared__ float sm[256];
  long long bc = blockIdx.x;
  const float* p = x + (bc << 16);
  float s = 0.f;
  for (int i = threadIdx.x; i < SP; i += 256) s += p[i];
  sm[threadIdx.x] = s;
  __syncthreads();
  for (int o = 128; o > 0; o >>= 1) {
    if (threadIdx.x < o) sm[threadIdx.x] += sm[threadIdx.x + o];
    __syncthreads();
  }
  if (threadIdx.x == 0) mean[bc] = sm[0] * (1.f / SP);
}

__global__ void se_fc_kernel(const float* __restrict__ mean, const float* __restrict__ f1,
                             const float* __restrict__ f2, float* __restrict__ sc)
{
  __shared__ float m[122];
  __shared__ float y1[7];
  int b = blockIdx.x, t = threadIdx.x;
  if (t < 122) m[t] = mean[b * 122 + t];
  __syncthreads();
  if (t < 7) {
    float s = 0.f;
    for (int c = 0; c < 122; ++c) s += m[c] * f1[t * 122 + c];
    y1[t] = fmaxf(s, 0.f);
  }
  __syncthreads();
  if (t < 122) {
    float s = 0.f;
    for (int j = 0; j < 7; ++j) s += y1[j] * f2[t * 7 + j];
    sc[b * 122 + t] = 1.f / (1.f + __expf(-s));
  }
}

__global__ void scale_ch_kernel(const float* __restrict__ x, const float* __restrict__ sc,
                                float* __restrict__ y, int C, long long n)
{
  long long i = (long long)blockIdx.x * blockDim.x + threadIdx.x;
  if (i >= n) return;
  long long bc = i >> 16;
  int c = (int)(bc % C);
  int b = (int)(bc / C);
  y[i] = x[i] * sc[b * C + c];
}

__global__ void soft_kernel(const float* __restrict__ Wx, const float* __restrict__ cmap,
                            const float* __restrict__ sigma, float* __restrict__ z, long long n)
{
  long long i = (long long)blockIdx.x * blockDim.x + threadIdx.x;
  if (i >= n) return;
  int sp = (int)(i & 65535);
  long long bc = i >> 16;
  int b = (int)(bc / 121);
  float t = cmap[i] * sigma[((long long)b << 16) + sp];
  float v = Wx[i];
  z[i] = v > t ? v - t : (v < -t ? v + t : 0.f);
}

__global__ void zero1_kernel(float* p) { *p = 0.f; }

__global__ void sqdiff_kernel(const float* __restrict__ a, const float* __restrict__ b,
                              float* __restrict__ out, long long n)
{
  __shared__ float sm[256];
  float s = 0.f;
  for (long long i = (long long)blockIdx.x * blockDim.x + threadIdx.x; i < n;
       i += (long long)gridDim.x * blockDim.x) {
    float d = a[i] - b[i];
    s += d * d;
  }
  sm[threadIdx.x] = s;
  __syncthreads();
  for (int o = 128; o > 0; o >>= 1) {
    if (threadIdx.x < o) sm[threadIdx.x] += sm[threadIdx.x + o];
    __syncthreads();
  }
  if (threadIdx.x == 0) atomicAdd(out, sm[0]);
}

// ======================================================================
// Host orchestration
// ======================================================================
static inline void launch_conv(hipStream_t st, const float* in, int inCtot, int inCoff,
                               const float* w, int wD1, int rot, const float* bias,
                               float* out, int outCtot,
                               int Cin, int Cout, int ks,
                               float scale, int clip, float lo, float hi)
{
  dim3 grid(8192, (Cout + 15) / 16);
  if (ks == 11) {
    if (rot)
      wmma_conv_t<11, true><<<grid, 32, 0, st>>>(in, inCtot, inCoff, w, wD1, bias,
                                                 out, outCtot, Cin, Cout, scale, clip, lo, hi);
    else
      wmma_conv_t<11, false><<<grid, 32, 0, st>>>(in, inCtot, inCoff, w, wD1, bias,
                                                  out, outCtot, Cin, Cout, scale, clip, lo, hi);
  } else if (ks == 3) {
    wmma_conv_t<3, false><<<grid, 32, 0, st>>>(in, inCtot, inCoff, w, wD1, bias,
                                               out, outCtot, Cin, Cout, scale, clip, lo, hi);
  } else {
    wmma_conv_t<1, false><<<grid, 32, 0, st>>>(in, inCtot, inCoff, w, wD1, bias,
                                               out, outCtot, Cin, Cout, scale, clip, lo, hi);
  }
}

static inline void run_tblock(hipStream_t st, const float* input, const float* sigma,
                              const float* Wx,
                              float* xcat, float* Lbuf, float* ha, float* hb,
                              float* smallb, float* Cbuf, float* zout,
                              const float* n1w, const float* n1b,
                              const float* thw, const float* tdw,
                              const float* fnw, const float* fnb, const float* pw,
                              const float* f1, const float* f2,
                              const float* cw, const float* cb)
{
  const long long n122 = 2ll * 122 * SP, n732 = 2ll * 732 * SP, n121 = 2ll * 121 * SP;
  concat_kernel<<<(int)((n122 + 255) / 256), 256, 0, st>>>(input, Wx, xcat, n122);
  ln_c_kernel<<<1024, 128, 0, st>>>(xcat, 122ll * SP, 0ll, (const float*)nullptr, 0ll, 0ll,
                                    Lbuf, 122ll * SP, 0ll, n1w, n1b, 122);
  // 1x1 conv 122 -> 732 (no bias)
  launch_conv(st, Lbuf, 122, 0, thw, 0, 0, nullptr, ha, 732, 122, 732, 1, 1.f, 0, 0.f, 0.f);
  dw3x3_kernel<<<(int)((n732 + 255) / 256), 256, 0, st>>>(ha, tdw, hb, 732, n732);
  // q*k 8x8 patch circular conv  (hb ch[0:244) x hb ch[244:488) -> ha ch[0:244))
  patch_cconv_kernel<<<2 * 244 * 1024, 64, 0, st>>>(hb, ha);
  // fsas = v * LN(o):  ha[0:244) LN'd, * hb[488:732) -> ha[244:488)
  ln_c_kernel<<<1024, 128, 0, st>>>(ha, 732ll * SP, 0ll, hb, 732ll * SP, 488ll * SP,
                                    ha, 732ll * SP, 244ll * SP, fnw, fnb, 244);
  // 1x1 conv 244 -> 122 (no bias)
  launch_conv(st, ha, 732, 244, pw, 0, 0, nullptr, Lbuf, 122, 244, 122, 1, 1.f, 0, 0.f, 0.f);
  // residual: out1 = xcat + fsas
  add_inplace_kernel<<<(int)((n122 + 255) / 256), 256, 0, st>>>(xcat, Lbuf, n122);
  // SE
  avgpool_kernel<<<244, 256, 0, st>>>(xcat, smallb);
  se_fc_kernel<<<2, 128, 0, st>>>(smallb, f1, f2, smallb + 256);
  scale_ch_kernel<<<(int)((n122 + 255) / 256), 256, 0, st>>>(xcat, smallb + 256, Lbuf, 122, n122);
  // c = clip(conv3x3(se_out) + bias, 0, 10)
  launch_conv(st, Lbuf, 122, 0, cw, 0, 0, cb, Cbuf, 121, 122, 121, 3, 1.f, 1, 0.f, 10.f);
  // z = soft(Wx, c * sigma)
  soft_kernel<<<(int)((n121 + 255) / 256), 256, 0, st>>>(Wx, Cbuf, sigma, zout, n121);
}

extern "C" void kernel_launch(void* const* d_in, const int* in_sizes, int n_in,
                              void* d_out, int out_size, void* d_ws, size_t ws_size,
                              hipStream_t stream)
{
  (void)in_sizes; (void)n_in; (void)out_size; (void)ws_size;
  const float* input = (const float*)d_in[0];
  const float* sigma = (const float*)d_in[1];
  const float* w1 = (const float*)d_in[2];
  const float* b1 = (const float*)d_in[3];
  const float* w2 = (const float*)d_in[4];
  const float* b2 = (const float*)d_in[5];
  const float* n1w = (const float*)d_in[6];
  const float* n1b = (const float*)d_in[7];
  const float* thw = (const float*)d_in[8];
  const float* tdw = (const float*)d_in[9];
  const float* fnw = (const float*)d_in[10];
  const float* fnb = (const float*)d_in[11];
  const float* pw  = (const float*)d_in[12];
  const float* se1_f1 = (const float*)d_in[13];
  const float* se1_f2 = (const float*)d_in[14];
  const float* se2_f1 = (const float*)d_in[15];
  const float* se2_f2 = (const float*)d_in[16];
  const float* c1w = (const float*)d_in[17];
  const float* c1b = (const float*)d_in[18];
  const float* c2w = (const float*)d_in[19];
  const float* c2b = (const float*)d_in[20];

  float* out = (float*)d_out;

  // ---- workspace layout (floats) : total ~287.3M floats = ~1.15 GB ----
  const size_t n121 = 2ull * 121 * SP;
  const size_t n122 = 2ull * 122 * SP;
  const size_t n732 = 2ull * 732 * SP;
  float* ws    = (float*)d_ws;
  float* Wx1   = ws;                 // persists whole run (needed for temp & loss)
  float* bufA  = Wx1  + n121;        // Wx2 -> Wt2 -> WTW2
  float* zbuf  = bufA + n121;        // z1 -> z2 -> temp
  float* Cbuf  = zbuf + n121;        // c1 / c2
  float* xcat  = Cbuf + n121;        // concat / out1
  float* Lbuf  = xcat + n122;        // ln1 / pw-out / se-out
  float* ha    = Lbuf + n122;        // 732-ch hidden (also o / fsas regions)
  float* hb    = ha   + n732;        // 732-ch depthwise output
  float* smallb = hb  + n732;        // SE mean (244) + SE scale (244)

  // Wx1 = conv_reflect(input, w1) + b1        [1 -> 121, 11x11]
  launch_conv(stream, input, 1, 0, w1, 0, 0, b1, Wx1, 121, 1, 121, 11, 1.f, 0, 0.f, 0.f);

  // iteration 1: tblock + SE1 + c1 + soft -> z1
  run_tblock(stream, input, sigma, Wx1, xcat, Lbuf, ha, hb, smallb, Cbuf, zbuf,
             n1w, n1b, thw, tdw, fnw, fnb, pw, se1_f1, se1_f2, c1w, c1b);

  // Wx2 = conv_reflect(z1, w2) + b2           [121 -> 121, 11x11]
  launch_conv(stream, zbuf, 121, 0, w2, 0, 0, b2, bufA, 121, 121, 121, 11, 1.f, 0, 0.f, 0.f);

  // iteration 2: tblock + SE2 + c2 + soft -> z2
  run_tblock(stream, input, sigma, bufA, xcat, Lbuf, ha, hb, smallb, Cbuf, zbuf,
             n1w, n1b, thw, tdw, fnw, fnb, pw, se2_f1, se2_f2, c2w, c2b);

  // Wt2 = conv_reflect(z2, rot180(w2)) / 121
  launch_conv(stream, zbuf, 121, 0, w2, 121, 1, nullptr, bufA, 121, 121, 121, 11,
              1.f / 121.f, 0, 0.f, 0.f);
  // Wt = conv_reflect(Wt2, rot180(w1)) / 121  -> d_out[0 : 131072)
  launch_conv(stream, bufA, 121, 0, w1, 1, 1, nullptr, out, 1, 121, 1, 11,
              1.f / 121.f, 0, 0.f, 0.f);

  // temp = conv_reflect(Wx1, w2) + b2
  launch_conv(stream, Wx1, 121, 0, w2, 0, 0, b2, zbuf, 121, 121, 121, 11, 1.f, 0, 0.f, 0.f);
  // WTW2 = conv_reflect(temp, rot180(w2)) / 121
  launch_conv(stream, zbuf, 121, 0, w2, 121, 1, nullptr, bufA, 121, 121, 121, 11,
              1.f / 121.f, 0, 0.f, 0.f);

  // W2loss = sum((WTW2 - Wx1)^2) -> d_out[131072 + 14641]
  float* lossp = out + 131072 + 14641;
  zero1_kernel<<<1, 1, 0, stream>>>(lossp);
  sqdiff_kernel<<<1024, 256, 0, stream>>>(bufA, Wx1, lossp, (long long)n121);

  // w1 pass-through -> d_out[131072 : 131072+14641)
  hipMemcpyAsync(out + 131072, w1, 14641 * sizeof(float), hipMemcpyDeviceToDevice, stream);
}
